// DRGRU_62216896250515
// MI455X (gfx1250) — compile-verified, matrix operand order checked by reading, not run on hardware
//
#include <hip/hip_runtime.h>
#include <hip/hip_bf16.h>
#include <math.h>

// ---------------- problem constants ----------------
#define B_ 8
#define N_ 1024
#define D_ 64
#define F_ 128          // 2*D
#define KP_ 512         // F*M
#define M_ 4

typedef __bf16 bf16;
typedef __attribute__((ext_vector_type(16))) __bf16 v16bf;
typedef __attribute__((ext_vector_type(8)))  __bf16 v8bf;
typedef __attribute__((ext_vector_type(8)))  float  v8f;

// element counts
#define SZ_A   ((size_t)B_ * N_ * N_)     // one A matrix (bf16 elems)
#define SZ_XT  ((size_t)B_ * F_ * N_)     // transposed feature matrix (bf16)
#define SZ_Y   ((size_t)B_ * N_ * F_)     // one diffusion result (f32)
#define SZ_S   ((size_t)B_ * N_ * KP_)    // one interleaved stack (bf16)
#define SZ_G   ((size_t)B_ * N_ * D_)     // r/u/c (f32)

// Load a 16-element bf16 WMMA fragment for lane (halfSel = lane>>4).
// ISA 16-bit A/B layout: e[0..7] = K = 8*half + 0..7 ; e[8..15] = K = 16 + 8*half + 0..7
__device__ __forceinline__ v16bf load_frag16(const bf16* rowk0, int halfSel) {
    const v8bf a0 = *(const v8bf*)(rowk0 + 8 * halfSel);
    const v8bf a1 = *(const v8bf*)(rowk0 + 16 + 8 * halfSel);
    v16bf r;
#pragma unroll
    for (int i = 0; i < 8; ++i) { r[i] = a0[i]; r[i + 8] = a1[i]; }
    return r;
}

// ---------------- row/col sums of adj (+1 for identity) ----------------
__global__ __launch_bounds__(256) void k_sums(const float* __restrict__ adj,
                                              float* __restrict__ rowsum,
                                              float* __restrict__ colsum) {
    const int b = blockIdx.y, i = blockIdx.x, t = threadIdx.x;
    const float* base = adj + (size_t)b * N_ * N_;
    float rs = 0.f, cs = 0.f;
    for (int j = t; j < N_; j += 256) {
        rs += base[(size_t)i * N_ + j];
        cs += base[(size_t)j * N_ + i];
    }
    __shared__ float sr[256], sc[256];
    sr[t] = rs; sc[t] = cs; __syncthreads();
    for (int s = 128; s > 0; s >>= 1) {
        if (t < s) { sr[t] += sr[t + s]; sc[t] += sc[t + s]; }
        __syncthreads();
    }
    if (t == 0) {
        rowsum[b * N_ + i] = sr[0] + 1.0f;
        colsum[b * N_ + i] = sc[0] + 1.0f;
    }
}

// ---------------- build A0 (direct) and A1 (LDS transpose), bf16 ----------------
__global__ __launch_bounds__(256) void k_norm(const float* __restrict__ adj,
                                              const float* __restrict__ rowsum,
                                              const float* __restrict__ colsum,
                                              bf16* __restrict__ A0,
                                              bf16* __restrict__ A1) {
    const int b = blockIdx.z, i0 = blockIdx.y * 32, j0 = blockIdx.x * 32;
    const int tx = threadIdx.x, ty = threadIdx.y;   // (32, 8)
    __shared__ float tile[32][33];
    const float* base = adj + (size_t)b * N_ * N_;
    bf16* A0b = A0 + (size_t)b * N_ * N_;
    bf16* A1b = A1 + (size_t)b * N_ * N_;
#pragma unroll
    for (int rr = 0; rr < 4; ++rr) {
        const int r = ty + rr * 8;
        const int i = i0 + r, j = j0 + tx;
        const float v = base[(size_t)i * N_ + j];
        tile[r][tx] = v;
        const float a0 = (v + (i == j ? 1.0f : 0.0f)) / colsum[b * N_ + i];
        A0b[(size_t)i * N_ + j] = (bf16)a0;
    }
    __syncthreads();
#pragma unroll
    for (int rr = 0; rr < 4; ++rr) {
        const int J = j0 + ty + rr * 8;   // output row of A1
        const int I = i0 + tx;            // output col of A1
        const float v = tile[tx][ty + rr * 8];   // adj[I][J]
        const float a1 = (v + (I == J ? 1.0f : 0.0f)) / rowsum[b * N_ + J];
        A1b[(size_t)J * N_ + I] = (bf16)a1;
    }
}

// ---------------- generic f32 -> bf16 convert ----------------
__global__ void k_cvt(const float* __restrict__ in, bf16* __restrict__ out, int n) {
    const int idx = blockIdx.x * blockDim.x + threadIdx.x;
    if (idx < n) out[idx] = (bf16)in[idx];
}

// ---------------- build XT[b][f][n] bf16 from inputs/hx (optionally r*hx) ----------------
__global__ void k_pack_xct(const float* __restrict__ inputs, const float* __restrict__ hx,
                           const float* __restrict__ rgate, int useR,
                           bf16* __restrict__ XT) {
    const size_t idx = (size_t)blockIdx.x * blockDim.x + threadIdx.x;
    if (idx >= SZ_XT) return;
    const int n = (int)(idx % N_);
    const size_t t = idx / N_;
    const int f = (int)(t % F_);
    const int b = (int)(t / F_);
    const size_t node = (size_t)b * N_ + n;
    float v;
    if (f < D_) v = inputs[node * D_ + f];
    else {
        v = hx[node * D_ + (f - D_)];
        if (useR) v *= rgate[node * D_ + (f - D_)];
    }
    XT[((size_t)b * F_ + f) * N_ + n] = (bf16)v;
}

// ---------------- diffusion GEMM: Y = A_bf16 @ X  (X given transposed bf16) ----------------
// grid: (N/16, B, 2 branches), block: 1 wave (32). Each wave: 16 nodes x 128 feats.
// All 8 B-fragments live in distinct registers so the load clause can run ahead of
// the WMMA stream (partial s_wait_loadcnt instead of a full drain per tile).
__global__ __launch_bounds__(32) void k_gemm_diff(const bf16* __restrict__ Ab,
                                                  const bf16* __restrict__ XTb,
                                                  size_t xBranchStride,
                                                  float* __restrict__ Yb,
                                                  bf16* __restrict__ YTb) {
    const int lane = threadIdx.x;
    const int halfSel = lane >> 4;
    const int lm = lane & 15;
    const int node0 = blockIdx.x * 16;
    const int b = blockIdx.y;
    const int branch = blockIdx.z;

    const bf16* A  = Ab  + (size_t)(branch * B_ + b) * N_ * N_;
    const bf16* XT = XTb + (size_t)branch * xBranchStride + (size_t)b * F_ * N_;
    float* Y  = Yb  + (size_t)(branch * B_ + b) * N_ * F_;
    bf16*  YT = YTb + (size_t)(branch * B_ + b) * F_ * N_;

    v8f acc[8];
#pragma unroll
    for (int ct = 0; ct < 8; ++ct)
#pragma unroll
        for (int i = 0; i < 8; ++i) acc[ct][i] = 0.0f;

    const bf16* Arow = A + (size_t)(node0 + lm) * N_;
    const bf16* Brow = XT + (size_t)lm * N_;

    for (int k0 = 0; k0 < N_; k0 += 32) {
        __builtin_prefetch(Arow + k0 + 64, 0, 3);   // near-cache prefetch on A stream
        const v16bf af = load_frag16(Arow + k0, halfSel);
        v16bf bfrag[8];
#pragma unroll
        for (int ct = 0; ct < 8; ++ct)
            bfrag[ct] = load_frag16(Brow + (size_t)(ct * 16) * N_ + k0, halfSel);
#pragma unroll
        for (int ct = 0; ct < 8; ++ct)
            acc[ct] = __builtin_amdgcn_wmma_f32_16x16x32_bf16(
                false, af, false, bfrag[ct], (short)0, acc[ct], false, false);
    }
    // C/D layout: VGPR r -> row = r + 8*half, col = lane%16
#pragma unroll
    for (int ct = 0; ct < 8; ++ct) {
#pragma unroll
        for (int r = 0; r < 8; ++r) {
            const int node = node0 + r + 8 * halfSel;
            const int feat = ct * 16 + lm;
            const float v = acc[ct][r];
            Y[(size_t)node * F_ + feat] = v;
            YT[(size_t)feat * N_ + node] = (bf16)v;
        }
    }
}

// ---------------- build interleaved stack s[node][4f+m] bf16 ----------------
// m=0 -> xc ; m=1,2 -> A x ; m=3 -> A^2 x
__global__ void k_pack_s(const float* __restrict__ inputs, const float* __restrict__ hx,
                         const float* __restrict__ rgate, int useR,
                         const float* __restrict__ Y1, const float* __restrict__ Y2,
                         bf16* __restrict__ S) {
    const size_t idx = (size_t)blockIdx.x * blockDim.x + threadIdx.x;
    if (idx >= SZ_S) return;
    const int k = (int)(idx % KP_);
    const size_t node = idx / KP_;
    const int f = k >> 2, m = k & 3;
    float v;
    if (m == 0) {
        if (f < D_) v = inputs[node * D_ + f];
        else {
            v = hx[node * D_ + (f - D_)];
            if (useR) v *= rgate[node * D_ + (f - D_)];
        }
    } else if (m == 3) v = Y2[node * F_ + f];
    else               v = Y1[node * F_ + f];
    S[node * KP_ + k] = (bf16)v;
}

// ---------------- projection GEMM: out = act(Wa@s0^T + Wb@s1^T + bA + bB) ----------------
// grid: (B*N/128, Mfeats/16), block: 1 wave. mode 0: sigmoid -> split r/u ; mode 1: tanh -> c
__global__ __launch_bounds__(32) void k_proj(const bf16* __restrict__ Wa,
                                             const bf16* __restrict__ Wb,
                                             const bf16* __restrict__ S0,
                                             const bf16* __restrict__ S1,
                                             const float* __restrict__ bA,
                                             const float* __restrict__ bB,
                                             float* __restrict__ out0,
                                             float* __restrict__ out1,
                                             int mode) {
    const int lane = threadIdx.x;
    const int halfSel = lane >> 4;
    const int lm = lane & 15;
    const size_t n0 = (size_t)blockIdx.x * 128;
    const int feat0 = blockIdx.y * 16;

    v8f acc[8];
#pragma unroll
    for (int ct = 0; ct < 8; ++ct)
#pragma unroll
        for (int i = 0; i < 8; ++i) acc[ct][i] = 0.0f;

    const bf16* WaRow = Wa + (size_t)(feat0 + lm) * KP_;
    const bf16* WbRow = Wb + (size_t)(feat0 + lm) * KP_;
    const bf16* S0row = S0 + (n0 + lm) * KP_;
    const bf16* S1row = S1 + (n0 + lm) * KP_;

    for (int k0 = 0; k0 < KP_; k0 += 32) {
        const v16bf af = load_frag16(WaRow + k0, halfSel);
        v16bf bfrag[8];
#pragma unroll
        for (int ct = 0; ct < 8; ++ct)
            bfrag[ct] = load_frag16(S0row + (size_t)(ct * 16) * KP_ + k0, halfSel);
#pragma unroll
        for (int ct = 0; ct < 8; ++ct)
            acc[ct] = __builtin_amdgcn_wmma_f32_16x16x32_bf16(
                false, af, false, bfrag[ct], (short)0, acc[ct], false, false);
    }
    for (int k0 = 0; k0 < KP_; k0 += 32) {
        const v16bf af = load_frag16(WbRow + k0, halfSel);
        v16bf bfrag[8];
#pragma unroll
        for (int ct = 0; ct < 8; ++ct)
            bfrag[ct] = load_frag16(S1row + (size_t)(ct * 16) * KP_ + k0, halfSel);
#pragma unroll
        for (int ct = 0; ct < 8; ++ct)
            acc[ct] = __builtin_amdgcn_wmma_f32_16x16x32_bf16(
                false, af, false, bfrag[ct], (short)0, acc[ct], false, false);
    }

#pragma unroll
    for (int ct = 0; ct < 8; ++ct) {
#pragma unroll
        for (int r = 0; r < 8; ++r) {
            const int feat = feat0 + r + 8 * halfSel;      // M row
            const size_t node = n0 + ct * 16 + lm;         // N col
            float v = acc[ct][r] + bA[feat] + bB[feat];
            if (mode == 0) {
                v = 1.0f / (1.0f + expf(-v));
                if (feat < D_) out0[node * D_ + feat] = v;
                else           out1[node * D_ + (feat - D_)] = v;
            } else {
                out0[node * D_ + feat] = tanhf(v);
            }
        }
    }
}

// ---------------- GRU combine ----------------
__global__ void k_final(const float* __restrict__ hx, const float* __restrict__ u,
                        const float* __restrict__ c, float* __restrict__ out, int n) {
    const int idx = blockIdx.x * blockDim.x + threadIdx.x;
    if (idx < n) {
        const float uu = u[idx];
        out[idx] = uu * hx[idx] + (1.0f - uu) * c[idx];
    }
}

static inline size_t alignup(size_t x) { return (x + 255) & ~(size_t)255; }

extern "C" void kernel_launch(void* const* d_in, const int* in_sizes, int n_in,
                              void* d_out, int out_size, void* d_ws, size_t ws_size,
                              hipStream_t stream) {
    (void)in_sizes; (void)n_in; (void)out_size; (void)ws_size;
    const float* inputs = (const float*)d_in[0];
    const float* hx     = (const float*)d_in[1];
    const float* adj    = (const float*)d_in[2];
    const float* W0     = (const float*)d_in[3];
    const float* b0     = (const float*)d_in[4];
    const float* W1     = (const float*)d_in[5];
    const float* b1     = (const float*)d_in[6];
    const float* Wc0    = (const float*)d_in[7];
    const float* bc0    = (const float*)d_in[8];
    const float* Wc1    = (const float*)d_in[9];
    const float* bc1    = (const float*)d_in[10];
    float* out = (float*)d_out;

    // -------- workspace carve-out --------
    char* w = (char*)d_ws;
    size_t off = 0;
    bf16* A0bf = (bf16*)(w + off); off += alignup(2 * SZ_A * sizeof(bf16));   // A1 follows A0
    bf16* XCT  = (bf16*)(w + off); off += alignup(SZ_XT * sizeof(bf16));
    float* Yf  = (float*)(w + off); off += alignup(4 * SZ_Y * sizeof(float)); // slots: y0,y1,y0b,y1b
    bf16* YT   = (bf16*)(w + off); off += alignup(4 * SZ_XT * sizeof(bf16));
    bf16* S0   = (bf16*)(w + off); off += alignup(2 * SZ_S * sizeof(bf16));   // S1 follows S0
    bf16* S1   = S0 + SZ_S;
    float* rbuf = (float*)(w + off); off += alignup(SZ_G * sizeof(float));
    float* ubuf = (float*)(w + off); off += alignup(SZ_G * sizeof(float));
    float* cbuf = (float*)(w + off); off += alignup(SZ_G * sizeof(float));
    float* rowsum = (float*)(w + off); off += alignup((size_t)B_ * N_ * sizeof(float));
    float* colsum = (float*)(w + off); off += alignup((size_t)B_ * N_ * sizeof(float));
    bf16* Wbf = (bf16*)(w + off); off += alignup((size_t)(2 * F_ + 2 * D_) * KP_ * sizeof(bf16));
    bf16* W0bf  = Wbf;
    bf16* W1bf  = Wbf + (size_t)F_ * KP_;
    bf16* Wc0bf = Wbf + (size_t)2 * F_ * KP_;
    bf16* Wc1bf = Wbf + (size_t)2 * F_ * KP_ + (size_t)D_ * KP_;

    // -------- graph normalization --------
    k_sums<<<dim3(N_, B_), 256, 0, stream>>>(adj, rowsum, colsum);
    k_norm<<<dim3(32, 32, B_), dim3(32, 8), 0, stream>>>(adj, rowsum, colsum,
                                                         A0bf, A0bf + SZ_A);
    // -------- weights to bf16 --------
    k_cvt<<<(F_ * KP_ + 255) / 256, 256, 0, stream>>>(W0, W0bf, F_ * KP_);
    k_cvt<<<(F_ * KP_ + 255) / 256, 256, 0, stream>>>(W1, W1bf, F_ * KP_);
    k_cvt<<<(D_ * KP_ + 255) / 256, 256, 0, stream>>>(Wc0, Wc0bf, D_ * KP_);
    k_cvt<<<(D_ * KP_ + 255) / 256, 256, 0, stream>>>(Wc1, Wc1bf, D_ * KP_);

    const int packXBlocks = (int)((SZ_XT + 255) / 256);
    const int packSBlocks = (int)((SZ_S + 255) / 256);

    // ================= gate phase =================
    k_pack_xct<<<packXBlocks, 256, 0, stream>>>(inputs, hx, rbuf, 0, XCT);
    // hop 1: y = A @ xc (both branches)
    k_gemm_diff<<<dim3(N_ / 16, B_, 2), 32, 0, stream>>>(A0bf, XCT, 0, Yf, YT);
    // hop 2: yb = A @ y
    k_gemm_diff<<<dim3(N_ / 16, B_, 2), 32, 0, stream>>>(A0bf, YT, SZ_XT,
                                                         Yf + 2 * SZ_Y, YT + 2 * SZ_XT);
    // interleaved stacks
    k_pack_s<<<packSBlocks, 256, 0, stream>>>(inputs, hx, rbuf, 0,
                                              Yf + 0 * SZ_Y, Yf + 2 * SZ_Y, S0);
    k_pack_s<<<packSBlocks, 256, 0, stream>>>(inputs, hx, rbuf, 0,
                                              Yf + 1 * SZ_Y, Yf + 3 * SZ_Y, S1);
    // gates: sigmoid(W0@s0 + W1@s1 + b0 + b1) -> r, u
    k_proj<<<dim3(B_ * N_ / 128, F_ / 16), 32, 0, stream>>>(W0bf, W1bf, S0, S1,
                                                            b0, b1, rbuf, ubuf, 0);

    // ================= candidate phase =================
    k_pack_xct<<<packXBlocks, 256, 0, stream>>>(inputs, hx, rbuf, 1, XCT);
    k_gemm_diff<<<dim3(N_ / 16, B_, 2), 32, 0, stream>>>(A0bf, XCT, 0, Yf, YT);
    k_gemm_diff<<<dim3(N_ / 16, B_, 2), 32, 0, stream>>>(A0bf, YT, SZ_XT,
                                                         Yf + 2 * SZ_Y, YT + 2 * SZ_XT);
    k_pack_s<<<packSBlocks, 256, 0, stream>>>(inputs, hx, rbuf, 1,
                                              Yf + 0 * SZ_Y, Yf + 2 * SZ_Y, S0);
    k_pack_s<<<packSBlocks, 256, 0, stream>>>(inputs, hx, rbuf, 1,
                                              Yf + 1 * SZ_Y, Yf + 3 * SZ_Y, S1);
    // candidate: tanh(Wc0@s0 + Wc1@s1 + bc0 + bc1) -> c
    k_proj<<<dim3(B_ * N_ / 128, D_ / 16), 32, 0, stream>>>(Wc0bf, Wc1bf, S0, S1,
                                                            bc0, bc1, cbuf, nullptr, 1);

    // ================= GRU combine =================
    k_final<<<(int)((SZ_G + 255) / 256), 256, 0, stream>>>(hx, ubuf, cbuf, out, (int)SZ_G);
}
